// CSA_54425825575482
// MI455X (gfx1250) — matrix operand order repaired and checked by reading, not compiled, not run on hardware
//
#include <hip/hip_runtime.h>
#include <hip/hip_bf16.h>
#include <math.h>

typedef __attribute__((ext_vector_type(16))) __bf16 v16bf;
typedef __attribute__((ext_vector_type(8)))  float  v8f;
typedef __attribute__((ext_vector_type(4)))  unsigned int v4u;
typedef __attribute__((ext_vector_type(8)))  int v8i;
typedef __attribute__((ext_vector_type(4)))  int v4i;

constexpr int kT = 2048, kHID = 2048, kQCD = 1024, kH = 16, kD = 128;
constexpr int kSW = 512, kM = 32, kTOPK = 32, kG = 4, kDG = 512, kNIH = 4, kCI = 64;
constexpr int kNB = kT / kM;                 // 64 blocks
constexpr float kEPS = 1e-6f;
constexpr float kNEG = -1e30f;
constexpr float kInvSqrtD = 0.08838834764831845f;   // 1/sqrt(128)

#if __has_builtin(__builtin_amdgcn_tensor_load_to_lds) && __has_builtin(__builtin_amdgcn_s_wait_tensorcnt)
#define USE_TDM 1
#else
#define USE_TDM 0
#endif

static __device__ __forceinline__ __bf16 f2bf(float x) {
  union { float f; unsigned u; } a; a.f = x;
  unsigned r = (a.u + 0x7FFFu + ((a.u >> 16) & 1u)) >> 16;
  union { unsigned short s; __bf16 b; } o; o.s = (unsigned short)r;
  return o.b;
}

// ---------------------------------------------------------------- convert
__global__ void convert_f32_bf16(const float* __restrict__ src,
                                 __bf16* __restrict__ dst, int n) {
  int i = blockIdx.x * 256 + threadIdx.x;
  if (i < n) dst[i] = f2bf(src[i]);
}

// ---------------------------------------------------------------- WMMA GEMM
// C(MxN,f32) = A(MxK,bf16,row-major,lda) * B(KxN,bf16,row-major,ldb)
// block tile 64x128, 8 waves (2x4), each wave -> 32x32 via 2x2 WMMA frags.
// A tile staged by the Tensor Data Mover (tensor_load_to_lds), B staged
// manually transposed so fragment pairs are contiguous 32-bit LDS loads.
// Requires: M multiple of 64, K multiple of 32, N multiple of 16 (true here).
__global__ __launch_bounds__(256) void gemm_bf16_wmma(
    const __bf16* __restrict__ A, const __bf16* __restrict__ B,
    float* __restrict__ C, int M, int N, int K, int lda, int ldb, int ldc)
{
  __shared__ __align__(16) __bf16 sA[64 * 32];
  __shared__ __align__(16) __bf16 sBt[128 * 32];   // transposed: [n][k]
  const int tid   = threadIdx.x;
  const int lane  = tid & 31;
  const int wave  = tid >> 5;
  const int m16   = lane & 15;
  const int lh    = lane >> 4;                 // lane half
  const int waveM = (wave >> 2) * 32;          // 0 / 32
  const int waveN = (wave & 3) * 32;           // 0..96
  const int blockM = blockIdx.x * 64;
  const int blockN = blockIdx.y * 128;

  v8f c00 = {0,0,0,0,0,0,0,0};
  v8f c01 = c00, c10 = c00, c11 = c00;

  const int aRow = tid >> 2;                   // 0..63
  const int aCol = (tid & 3) * 8;              // 0,8,16,24
  const int bK   = tid >> 3;                   // 0..31
  const int bN0  = (tid & 7) * 16;             // 0..112

#if USE_TDM
  const unsigned ldsA = (unsigned)(uintptr_t)(void*)&sA[0];   // flat LDS addr[31:0]
  const unsigned long long gA0 =
      (unsigned long long)(uintptr_t)(A + (size_t)blockM * lda);
#endif

  for (int kk = 0; kk < K; kk += 32) {
#if USE_TDM
    if (wave == 0) {   // one TDM op per workgroup per K-step
      unsigned long long ga = gA0 + (unsigned long long)kk * 2ull;
      v4u g0;
      g0[0] = 1u;                                   // count=1, user mode
      g0[1] = ldsA;                                 // lds_addr (bytes)
      g0[2] = (unsigned)ga;                         // global_addr[31:0]
      g0[3] = (unsigned)(ga >> 32) | (2u << 30);    // global_addr[56:32] | type=2
      v8i g1;
      g1[0] = 1 << 16;                              // data_size = 2 bytes
      g1[1] = 32 << 16;                             // tensor_dim0 = 32
      g1[2] = 64 << 16;                             // tensor_dim1 = 64
      g1[3] = 32 << 16;                             // tile_dim0 = 32
      g1[4] = 64;                                   // tile_dim1 = 64, tile_dim2 = 0
      g1[5] = lda;                                  // tensor_dim0_stride[31:0]
      g1[6] = 0;                                    // stride hi / dim1_stride lo
      g1[7] = 0;
      v4i g2 = {0, 0, 0, 0};
      v4i g3 = {0, 0, 0, 0};
      v8i g4 = {0, 0, 0, 0, 0, 0, 0, 0};
      __builtin_amdgcn_tensor_load_to_lds(g0, g1, g2, g3, g4, 0);
    }
#else
    { // stage A 64x32 manually
      uint4 v = make_uint4(0u, 0u, 0u, 0u);
      int gm = blockM + aRow;
      if (gm < M) v = *(const uint4*)(A + (size_t)gm * lda + kk + aCol);
      *(uint4*)(sA + aRow * 32 + aCol) = v;
    }
#endif
    { // stage B 32x128 transposed; N is a multiple of 16 -> chunk predication
      int gk = kk + bK;
      const __bf16* bp = B + (size_t)gk * ldb + blockN + bN0;
      union { uint4 v; __bf16 hh[8]; } u0, u1;
      if (blockN + bN0 < N) {
        u0.v = *(const uint4*)(bp);
        u1.v = *(const uint4*)(bp + 8);
      } else {
        u0.v = make_uint4(0u, 0u, 0u, 0u);
        u1.v = u0.v;
      }
      #pragma unroll
      for (int j = 0; j < 8; ++j) sBt[(bN0 + j) * 32 + bK] = u0.hh[j];
      #pragma unroll
      for (int j = 0; j < 8; ++j) sBt[(bN0 + 8 + j) * 32 + bK] = u1.hh[j];
    }
    // prefetch next K-step tiles into cache while we compute this one
    if (kk + 32 < K) {
      if (blockM + aRow < M)
        __builtin_prefetch(A + (size_t)(blockM + aRow) * lda + kk + 32 + aCol, 0, 1);
      if (blockN + bN0 < N)
        __builtin_prefetch(B + (size_t)(kk + 32 + bK) * ldb + blockN + bN0, 0, 1);
    }
#if USE_TDM
    if (wave == 0) __builtin_amdgcn_s_wait_tensorcnt(0);
#endif
    __syncthreads();

    v16bf a0, a1, b0, b1;
    #pragma unroll
    for (int j = 0; j < 8; ++j) {
      // A-fragment K mapping (ISA 7.12.2, 16-bit A 16x32)
      int kA = (j < 4) ? (2 * j + 8 * lh) : (16 + 2 * (j - 4) + 8 * lh);
      union { unsigned u; __bf16 hh[2]; } u;
      u.u = *(const unsigned*)(sA + (waveM + m16) * 32 + kA);
      a0[2 * j] = u.hh[0]; a0[2 * j + 1] = u.hh[1];
      u.u = *(const unsigned*)(sA + (waveM + 16 + m16) * 32 + kA);
      a1[2 * j] = u.hh[0]; a1[2 * j + 1] = u.hh[1];
      // B-fragment: same K mapping as A
      int kB = (j < 4) ? (2 * j + 8 * lh) : (16 + 2 * (j - 4) + 8 * lh);
      u.u = *(const unsigned*)(sBt + (waveN + m16) * 32 + kB);
      b0[2 * j] = u.hh[0]; b0[2 * j + 1] = u.hh[1];
      u.u = *(const unsigned*)(sBt + (waveN + 16 + m16) * 32 + kB);
      b1[2 * j] = u.hh[0]; b1[2 * j + 1] = u.hh[1];
    }
    c00 = __builtin_amdgcn_wmma_f32_16x16x32_bf16(false, a0, false, b0, (short)0, c00, false, false);
    c01 = __builtin_amdgcn_wmma_f32_16x16x32_bf16(false, a0, false, b1, (short)0, c01, false, false);
    c10 = __builtin_amdgcn_wmma_f32_16x16x32_bf16(false, a1, false, b0, (short)0, c10, false, false);
    c11 = __builtin_amdgcn_wmma_f32_16x16x32_bf16(false, a1, false, b1, (short)0, c11, false, false);
    __syncthreads();
  }
  #pragma unroll
  for (int r = 0; r < 8; ++r) {
    int gm0 = blockM + waveM + r + 8 * lh;
    int gm1 = gm0 + 16;
    int gn0 = blockN + waveN + m16;
    int gn1 = gn0 + 16;
    if (gm0 < M && gn0 < N) C[(size_t)gm0 * ldc + gn0] = c00[r];
    if (gm0 < M && gn1 < N) C[(size_t)gm0 * ldc + gn1] = c01[r];
    if (gm1 < M && gn0 < N) C[(size_t)gm1 * ldc + gn0] = c10[r];
    if (gm1 < M && gn1 < N) C[(size_t)gm1 * ldc + gn1] = c11[r];
  }
}

// ---------------------------------------------------------------- q: rope then rms
__global__ __launch_bounds__(128) void q_rope_rms(
    const float* __restrict__ qflat, const float* __restrict__ qw,
    float* __restrict__ qout, __bf16* __restrict__ qbf)
{
  int t = blockIdx.x, h = blockIdx.y, d = threadIdx.x;
  __shared__ float buf[128];
  __shared__ float red[4];
  float x = qflat[(size_t)t * (kH * kD) + h * kD + d];
  buf[d] = x;
  __syncthreads();
  float y;
  if (d < 32) {
    float inv = __powf(10000.0f, -(float)(2 * d) / 64.0f);
    float ang = (float)t * inv;
    y = buf[d] * __cosf(ang) - buf[d + 32] * __sinf(ang);
  } else if (d < 64) {
    int i = d - 32;
    float inv = __powf(10000.0f, -(float)(2 * i) / 64.0f);
    float ang = (float)t * inv;
    y = buf[d] * __cosf(ang) + buf[i] * __sinf(ang);
  } else y = x;
  float ss = y * y;
  ss += __shfl_xor(ss, 1, 32);  ss += __shfl_xor(ss, 2, 32);
  ss += __shfl_xor(ss, 4, 32);  ss += __shfl_xor(ss, 8, 32);
  ss += __shfl_xor(ss, 16, 32);
  if ((d & 31) == 0) red[d >> 5] = ss;
  __syncthreads();
  float tot = red[0] + red[1] + red[2] + red[3];
  float r = rsqrtf(tot / 128.0f + kEPS);
  float outv = y * r * qw[d];
  size_t o = ((size_t)h * kT + t) * kD + d;
  qout[o] = outv;
  qbf[o]  = f2bf(outv);
}

// ---------------------------------------------------------------- k: rms then rope
__global__ __launch_bounds__(128) void k_rms_rope(
    const float* __restrict__ kin, const float* __restrict__ kw,
    __bf16* __restrict__ kbf)
{
  int t = blockIdx.x, d = threadIdx.x;
  __shared__ float buf[128];
  __shared__ float red[4];
  float x = kin[(size_t)t * 128 + d];
  float ss = x * x;
  ss += __shfl_xor(ss, 1, 32);  ss += __shfl_xor(ss, 2, 32);
  ss += __shfl_xor(ss, 4, 32);  ss += __shfl_xor(ss, 8, 32);
  ss += __shfl_xor(ss, 16, 32);
  if ((d & 31) == 0) red[d >> 5] = ss;
  __syncthreads();
  float tot = red[0] + red[1] + red[2] + red[3];
  float r = rsqrtf(tot / 128.0f + kEPS);
  float xn = x * r * kw[d];
  buf[d] = xn;
  __syncthreads();
  float y;
  if (d < 32) {
    float inv = __powf(10000.0f, -(float)(2 * d) / 64.0f);
    float ang = (float)t * inv;
    y = buf[d] * __cosf(ang) - buf[d + 32] * __sinf(ang);
  } else if (d < 64) {
    int i = d - 32;
    float inv = __powf(10000.0f, -(float)(2 * i) / 64.0f);
    float ang = (float)t * inv;
    y = buf[d] * __cosf(ang) + buf[i] * __sinf(ang);
  } else y = xn;
  kbf[(size_t)t * 128 + d] = f2bf(y);
}

// ---------------------------------------------------------------- row rms (comp_kv)
__global__ __launch_bounds__(128) void rms_rows128(
    const float* __restrict__ in, const float* __restrict__ w, float* __restrict__ out)
{
  int row = blockIdx.x, d = threadIdx.x;
  __shared__ float red[4];
  float x = in[(size_t)row * 128 + d];
  float ss = x * x;
  ss += __shfl_xor(ss, 1, 32);  ss += __shfl_xor(ss, 2, 32);
  ss += __shfl_xor(ss, 4, 32);  ss += __shfl_xor(ss, 8, 32);
  ss += __shfl_xor(ss, 16, 32);
  if ((d & 31) == 0) red[d >> 5] = ss;
  __syncthreads();
  float tot = red[0] + red[1] + red[2] + red[3];
  float r = rsqrtf(tot / 128.0f + kEPS);
  out[(size_t)row * 128 + d] = x * r * w[d];
}

// ---------------------------------------------------------------- compress (softmax over M=32)
__global__ void compress_blocks(const float* __restrict__ cbuf, const float* __restrict__ zbuf,
                                const float* __restrict__ bias, float* __restrict__ out, int C)
{
  int n = blockIdx.x, ch = threadIdx.x;
  if (ch >= C) return;
  float zmax = kNEG;
  for (int m = 0; m < kM; ++m) {
    float z = zbuf[(size_t)(n * kM + m) * C + ch] + bias[m * C + ch];
    zmax = fmaxf(zmax, z);
  }
  float s = 0.0f, acc = 0.0f;
  for (int m = 0; m < kM; ++m) {
    float z = zbuf[(size_t)(n * kM + m) * C + ch] + bias[m * C + ch];
    float w = __expf(z - zmax);
    s += w;
    acc += w * cbuf[(size_t)(n * kM + m) * C + ch];
  }
  out[(size_t)n * C + ch] = acc / s;
}

// ---------------------------------------------------------------- gating weights h @ w_w
__global__ void wgt_kernel(const float* __restrict__ hin, const float* __restrict__ w_w,
                           float* __restrict__ wgt)
{
  int i = blockIdx.x * 256 + threadIdx.x;
  if (i >= kT * kNIH) return;
  int t = i >> 2, j = i & 3;
  float acc = 0.0f;
  for (int k = 0; k < kHID; ++k) acc += hin[(size_t)t * kHID + k] * w_w[k * kNIH + j];
  wgt[i] = acc;
}

// ---------------------------------------------------------------- indexer scores + top-k
__global__ __launch_bounds__(64) void select_topk(
    const float* __restrict__ qi, const float* __restrict__ kci,
    const float* __restrict__ wgt, int* __restrict__ top)
{
  int t = blockIdx.x, n = threadIdx.x;
  __shared__ float isc[kNB];
  __shared__ float sq[kNIH * kCI];
  __shared__ float swt[kNIH];
  for (int i = n; i < kNIH * kCI; i += 64) sq[i] = qi[(size_t)t * kNIH * kCI + i];
  if (n < kNIH) swt[n] = wgt[t * kNIH + n];
  __syncthreads();
  float acc = 0.0f;
  #pragma unroll
  for (int ih = 0; ih < kNIH; ++ih) {
    float d = 0.0f;
    for (int c = 0; c < kCI; ++c) d += sq[ih * kCI + c] * kci[(size_t)n * kCI + c];
    acc += swt[ih] * fmaxf(d, 0.0f);
  }
  bool valid = (n * kM + (kM - 1)) < t;
  isc[n] = valid ? acc : kNEG;
  __syncthreads();
  if (n == 0) {
    for (int k2 = 0; k2 < kTOPK; ++k2) {
      float best = -3e30f; int bj = -1;
      for (int j = 0; j < kNB; ++j) if (isc[j] > best) { best = isc[j]; bj = j; }
      if (best <= -1e29f) top[t * kTOPK + k2] = -1;
      else { top[t * kTOPK + k2] = bj; isc[bj] = -3e30f; }
    }
  }
}

// ---------------------------------------------------------------- sparse attention (32 picked blocks)
__global__ __launch_bounds__(128) void sparse_attn(
    const float* __restrict__ q, const float* __restrict__ kvn,
    const int* __restrict__ top, float* __restrict__ attn)
{
  int t = blockIdx.x, h = blockIdx.y, tid = threadIdx.x;
  __shared__ float sq[128];
  __shared__ float sp[kTOPK];
  __shared__ int   sidx[kTOPK];
  sq[tid] = q[((size_t)h * kT + t) * kD + tid];
  if (tid < kTOPK) sidx[tid] = top[t * kTOPK + tid];
  __syncthreads();
  if (tid < kTOPK) {
    int idx = sidx[tid];
    float sc = kNEG;
    if (idx >= 0) {
      float d = 0.0f;
      for (int f = 0; f < kD; ++f) d += sq[f] * kvn[(size_t)idx * kD + f];
      sc = d * kInvSqrtD;
    }
    float mx = sc;
    mx = fmaxf(mx, __shfl_xor(mx, 1, 32));  mx = fmaxf(mx, __shfl_xor(mx, 2, 32));
    mx = fmaxf(mx, __shfl_xor(mx, 4, 32));  mx = fmaxf(mx, __shfl_xor(mx, 8, 32));
    mx = fmaxf(mx, __shfl_xor(mx, 16, 32));
    float e = (sc > -1e29f) ? __expf(sc - mx) : 0.0f;
    float s = e;
    s += __shfl_xor(s, 1, 32);  s += __shfl_xor(s, 2, 32);
    s += __shfl_xor(s, 4, 32);  s += __shfl_xor(s, 8, 32);
    s += __shfl_xor(s, 16, 32);
    sp[tid] = (s > 0.0f) ? (e / s) : 0.0f;
  }
  __syncthreads();
  float acc = 0.0f;
  for (int k2 = 0; k2 < kTOPK; ++k2) {
    int idx = sidx[k2];
    if (idx >= 0) acc += sp[k2] * kvn[(size_t)idx * kD + tid];
  }
  attn[((size_t)t * kH + h) * kD + tid] = acc;
}

// ---------------------------------------------------------------- sliding-window flash attention
// 1 wave per (head, 16-query tile). WMMA for QK^T and PV, online softmax.
__global__ __launch_bounds__(32) void sw_attn(
    const __bf16* __restrict__ qbf,   // (H,T,D)
    const __bf16* __restrict__ kbf,   // (T,D)
    const __bf16* __restrict__ vbf,   // (T,D)
    const float* __restrict__ sink_logit,
    float* __restrict__ attn)         // (T,H,D) accumulate-add
{
  const int h = blockIdx.y;
  const int tBase = blockIdx.x * 16;
  const int lane = threadIdx.x;
  const int m16 = lane & 15;
  const int lh = lane >> 4;

  __shared__ __align__(16) __bf16 sK[32 * 128];     // [key][f]
  __shared__ __align__(16) __bf16 sVt[128 * 32];    // [f][key]
  __shared__ __align__(16) float  sP[16 * 32];

  // Q fragments: 16x128 split into 4 A-frags of K=32
  v16bf aq[4];
  {
    const __bf16* qrow = qbf + ((size_t)h * kT + tBase + m16) * kD;
    #pragma unroll
    for (int c = 0; c < 4; ++c) {
      #pragma unroll
      for (int j = 0; j < 8; ++j) {
        int kA = 32 * c + ((j < 4) ? (2 * j + 8 * lh) : (16 + 2 * (j - 4) + 8 * lh));
        union { unsigned u; __bf16 hh[2]; } u;
        u.u = *(const unsigned*)(qrow + kA);
        aq[c][2 * j] = u.hh[0]; aq[c][2 * j + 1] = u.hh[1];
      }
    }
  }

  const v8f vz = {0,0,0,0,0,0,0,0};
  v8f oacc[8];
  #pragma unroll
  for (int c = 0; c < 8; ++c) oacc[c] = vz;
  float mrow[8], lrow[8];
  #pragma unroll
  for (int r = 0; r < 8; ++r) { mrow[r] = kNEG; lrow[r] = 0.0f; }

  int sStart = tBase - (kSW - 1); if (sStart < 0) sStart = 0;
  const int kb0 = sStart >> 5;
  const int kb1 = (tBase + 15) >> 5;

  for (int kb = kb0; kb <= kb1; ++kb) {
    const int s0 = kb * 32;
    { // stage K row-major, V transposed
      int s = s0 + lane; if (s >= kT) s = kT - 1;
      const uint4* ksrc = (const uint4*)(kbf + (size_t)s * kD);
      uint4* kdst = (uint4*)(sK + lane * kD);
      #pragma unroll
      for (int q = 0; q < 16; ++q) kdst[q] = ksrc[q];
      const uint4* vsrc = (const uint4*)(vbf + (size_t)s * kD);
      #pragma unroll
      for (int q = 0; q < 16; ++q) {
        union { uint4 v; __bf16 hh[8]; } u; u.v = vsrc[q];
        #pragma unroll
        for (int j = 0; j < 8; ++j) sVt[(q * 8 + j) * 32 + lane] = u.hh[j];
      }
    }
    __syncthreads();

    // S = Q K^T : two 16x16 C frags over 4 K-chunks
    v8f sc0 = vz, sc1 = vz;
    #pragma unroll
    for (int c = 0; c < 4; ++c) {
      v16bf b0, b1;
      #pragma unroll
      for (int j = 0; j < 8; ++j) {
        int kB = 32 * c + ((j < 4) ? (2 * j + 8 * lh) : (16 + 2 * (j - 4) + 8 * lh));
        union { unsigned u; __bf16 hh[2]; } u;
        u.u = *(const unsigned*)(sK + m16 * kD + kB);
        b0[2 * j] = u.hh[0]; b0[2 * j + 1] = u.hh[1];
        u.u = *(const unsigned*)(sK + (16 + m16) * kD + kB);
        b1[2 * j] = u.hh[0]; b1[2 * j + 1] = u.hh[1];
      }
      sc0 = __builtin_amdgcn_wmma_f32_16x16x32_bf16(false, aq[c], false, b0, (short)0, sc0, false, false);
      sc1 = __builtin_amdgcn_wmma_f32_16x16x32_bf16(false, aq[c], false, b1, (short)0, sc1, false, false);
    }

    // online softmax in C-frag layout: row m = r + 8*lh, col n = m16 (+16)
    #pragma unroll
    for (int r = 0; r < 8; ++r) {
      int tq = tBase + r + 8 * lh;
      int si0 = s0 + m16;
      int si1 = s0 + 16 + m16;
      float v0 = sc0[r] * kInvSqrtD;
      float v1 = sc1[r] * kInvSqrtD;
      if (si0 > tq || (tq - si0) >= kSW) v0 = kNEG;
      if (si1 > tq || (tq - si1) >= kSW) v1 = kNEG;
      float mx = fmaxf(v0, v1);
      mx = fmaxf(mx, __shfl_xor(mx, 1, 32));  mx = fmaxf(mx, __shfl_xor(mx, 2, 32));
      mx = fmaxf(mx, __shfl_xor(mx, 4, 32));  mx = fmaxf(mx, __shfl_xor(mx, 8, 32));
      float mnew = fmaxf(mrow[r], mx);
      float p0 = 0.0f, p1 = 0.0f, corr = 1.0f;
      if (mnew > -1e29f) {
        corr = __expf(mrow[r] - mnew);
        p0 = (v0 > -1e29f) ? __expf(v0 - mnew) : 0.0f;
        p1 = (v1 > -1e29f) ? __expf(v1 - mnew) : 0.0f;
      }
      float ps = p0 + p1;
      ps += __shfl_xor(ps, 1, 32);  ps += __shfl_xor(ps, 2, 32);
      ps += __shfl_xor(ps, 4, 32);  ps += __shfl_xor(ps, 8, 32);
      lrow[r] = lrow[r] * corr + ps;
      mrow[r] = mnew;
      #pragma unroll
      for (int c2 = 0; c2 < 8; ++c2) oacc[c2][r] *= corr;
      sP[(r + 8 * lh) * 32 + m16] = p0;
      sP[(r + 8 * lh) * 32 + 16 + m16] = p1;
    }
    __syncthreads();

    // P as A-fragment (16x32 bf16), then PV over 8 output chunks of 16
    v16bf ap;
    #pragma unroll
    for (int j = 0; j < 8; ++j) {
      int kA = (j < 4) ? (2 * j + 8 * lh) : (16 + 2 * (j - 4) + 8 * lh);
      ap[2 * j]     = f2bf(sP[m16 * 32 + kA]);
      ap[2 * j + 1] = f2bf(sP[m16 * 32 + kA + 1]);
    }
    #pragma unroll
    for (int c = 0; c < 8; ++c) {
      v16bf bv;
      #pragma unroll
      for (int j = 0; j < 8; ++j) {
        int kB = (j < 4) ? (2 * j + 8 * lh) : (16 + 2 * (j - 4) + 8 * lh);
        union { unsigned u; __bf16 hh[2]; } u;
        u.u = *(const unsigned*)(sVt + (c * 16 + m16) * 32 + kB);
        bv[2 * j] = u.hh[0]; bv[2 * j + 1] = u.hh[1];
      }
      oacc[c] = __builtin_amdgcn_wmma_f32_16x16x32_bf16(false, ap, false, bv, (short)0, oacc[c], false, false);
    }
    __syncthreads();
  }

  // sink logit joins the denominator only
  float sl = sink_logit[h];
  float inv[8];
  #pragma unroll
  for (int r = 0; r < 8; ++r) {
    float denom = lrow[r] + __expf(sl - mrow[r]);
    inv[r] = 1.0f / denom;
  }
  #pragma unroll
  for (int c = 0; c < 8; ++c) {
    #pragma unroll
    for (int r = 0; r < 8; ++r) {
      int tq = tBase + r + 8 * lh;
      size_t oidx = ((size_t)tq * kH + h) * kD + c * 16 + m16;
      attn[oidx] += oacc[c][r] * inv[r];
    }
  }
}

// ================================================================ host
extern "C" void kernel_launch(void* const* d_in, const int* in_sizes, int n_in,
                              void* d_out, int out_size, void* d_ws, size_t ws_size,
                              hipStream_t stream)
{
  (void)in_sizes; (void)n_in; (void)out_size; (void)ws_size;
  const float* h_in      = (const float*)d_in[0];
  const float* w_qc      = (const float*)d_in[1];
  const float* w_qup     = (const float*)d_in[2];
  const float* w_kv_b    = (const float*)d_in[3];
  const float* w_z_b     = (const float*)d_in[4];
  const float* bias_b    = (const float*)d_in[5];
  const float* w_k       = (const float*)d_in[6];
  const float* w_v       = (const float*)d_in[7];
  const float* wi_kv     = (const float*)d_in[8];
  const float* wi_z      = (const float*)d_in[9];
  const float* bias_i    = (const float*)d_in[10];
  const float* w_dq      = (const float*)d_in[11];
  const float* w_iuq     = (const float*)d_in[12];
  const float* w_w       = (const float*)d_in[13];
  const float* q_norm_w  = (const float*)d_in[14];
  const float* k_norm_w  = (const float*)d_in[15];
  const float* sink_lg   = (const float*)d_in[16];
  const float* group_w   = (const float*)d_in[17];
  const float* final_w   = (const float*)d_in[18];

  char* ws = (char*)d_ws;
  size_t off = 0;
  auto alloc = [&](size_t bytes) -> void* {
    void* p = ws + off;
    off = (off + bytes + 255) & ~(size_t)255;
    return p;
  };

  __bf16* hbf    = (__bf16*)alloc((size_t)kT * kHID * 2);
  __bf16* wqc    = (__bf16*)alloc((size_t)kHID * kQCD * 2);
  float*  qc     = (float*) alloc((size_t)kT * kQCD * 4);
  __bf16* qcbf   = (__bf16*)alloc((size_t)kT * kQCD * 2);
  __bf16* wqup   = (__bf16*)alloc((size_t)kQCD * kH * kD * 2);
  float*  qflat  = (float*) alloc((size_t)kT * kH * kD * 4);
  float*  qn     = (float*) alloc((size_t)kH * kT * kD * 4);
  __bf16* qbf    = (__bf16*)alloc((size_t)kH * kT * kD * 2);
  __bf16* wkvb   = (__bf16*)alloc((size_t)kHID * kD * 2);
  __bf16* wzb    = (__bf16*)alloc((size_t)kHID * kD * 2);
  float*  kvb    = (float*) alloc((size_t)kT * kD * 4);
  float*  zb     = (float*) alloc((size_t)kT * kD * 4);
  float*  ckv    = (float*) alloc((size_t)kNB * kD * 4);
  float*  kvnrm  = (float*) alloc((size_t)kNB * kD * 4);
  __bf16* wik    = (__bf16*)alloc((size_t)kHID * kCI * 2);
  __bf16* wiz    = (__bf16*)alloc((size_t)kHID * kCI * 2);
  float*  cikv   = (float*) alloc((size_t)kT * kCI * 4);
  float*  ciz    = (float*) alloc((size_t)kT * kCI * 4);
  float*  kci    = (float*) alloc((size_t)kNB * kCI * 4);
  __bf16* wdq    = (__bf16*)alloc((size_t)kHID * kCI * 2);
  float*  dq     = (float*) alloc((size_t)kT * kCI * 4);
  __bf16* dqbf   = (__bf16*)alloc((size_t)kT * kCI * 2);
  __bf16* wiuq   = (__bf16*)alloc((size_t)kCI * kNIH * kCI * 2);
  float*  qi     = (float*) alloc((size_t)kT * kNIH * kCI * 4);
  float*  wgt    = (float*) alloc((size_t)kT * kNIH * 4);
  int*    topi   = (int*)   alloc((size_t)kT * kTOPK * 4);
  float*  attn   = (float*) alloc((size_t)kT * kH * kD * 4);
  __bf16* wk     = (__bf16*)alloc((size_t)kHID * kD * 2);
  __bf16* wv     = (__bf16*)alloc((size_t)kHID * kD * 2);
  float*  swk0   = (float*) alloc((size_t)kT * kD * 4);
  float*  swv    = (float*) alloc((size_t)kT * kD * 4);
  __bf16* kbf    = (__bf16*)alloc((size_t)kT * kD * 2);
  __bf16* vbf    = (__bf16*)alloc((size_t)kT * kD * 2);
  __bf16* attnbf = (__bf16*)alloc((size_t)kT * kH * kD * 2);
  __bf16* gw     = (__bf16*)alloc((size_t)kG * 512 * kDG * 2);
  float*  og     = (float*) alloc((size_t)kT * kG * kDG * 4);
  __bf16* ogbf   = (__bf16*)alloc((size_t)kT * kG * kDG * 2);
  __bf16* fw     = (__bf16*)alloc((size_t)kG * kDG * kHID * 2);

  auto conv = [&](const float* src, __bf16* dst, int n) {
    convert_f32_bf16<<<dim3((n + 255) / 256), dim3(256), 0, stream>>>(src, dst, n);
  };
  auto gemm = [&](const __bf16* A, const __bf16* B, float* C,
                  int M, int N, int K, int lda, int ldb, int ldc) {
    dim3 g((M + 63) / 64, (N + 127) / 128);
    gemm_bf16_wmma<<<g, dim3(256), 0, stream>>>(A, B, C, M, N, K, lda, ldb, ldc);
  };

  // ---- q path ----
  conv(h_in, hbf, kT * kHID);
  conv(w_qc, wqc, kHID * kQCD);
  gemm(hbf, wqc, qc, kT, kQCD, kHID, kHID, kQCD, kQCD);
  conv(qc, qcbf, kT * kQCD);
  conv(w_qup, wqup, kQCD * kH * kD);
  gemm(qcbf, wqup, qflat, kT, kH * kD, kQCD, kQCD, kH * kD, kH * kD);
  q_rope_rms<<<dim3(kT, kH), dim3(128), 0, stream>>>(qflat, q_norm_w, qn, qbf);

  // ---- compressed KV (big) ----
  conv(w_kv_b, wkvb, kHID * kD);
  conv(w_z_b, wzb, kHID * kD);
  gemm(hbf, wkvb, kvb, kT, kD, kHID, kHID, kD, kD);
  gemm(hbf, wzb,  zb,  kT, kD, kHID, kHID, kD, kD);
  compress_blocks<<<dim3(kNB), dim3(kD), 0, stream>>>(kvb, zb, bias_b, ckv, kD);
  rms_rows128<<<dim3(kNB), dim3(kD), 0, stream>>>(ckv, k_norm_w, kvnrm);

  // ---- indexer path ----
  conv(wi_kv, wik, kHID * kCI);
  conv(wi_z, wiz, kHID * kCI);
  gemm(hbf, wik, cikv, kT, kCI, kHID, kHID, kCI, kCI);
  gemm(hbf, wiz, ciz,  kT, kCI, kHID, kHID, kCI, kCI);
  compress_blocks<<<dim3(kNB), dim3(kCI), 0, stream>>>(cikv, ciz, bias_i, kci, kCI);
  conv(w_dq, wdq, kHID * kCI);
  gemm(hbf, wdq, dq, kT, kCI, kHID, kHID, kCI, kCI);
  conv(dq, dqbf, kT * kCI);
  conv(w_iuq, wiuq, kCI * kNIH * kCI);
  gemm(dqbf, wiuq, qi, kT, kNIH * kCI, kCI, kCI, kNIH * kCI, kNIH * kCI);
  wgt_kernel<<<dim3((kT * kNIH + 255) / 256), dim3(256), 0, stream>>>(h_in, w_w, wgt);
  select_topk<<<dim3(kT), dim3(64), 0, stream>>>(qi, kci, wgt, topi);

  // ---- sparse attention (writes attn) ----
  sparse_attn<<<dim3(kT, kH), dim3(128), 0, stream>>>(qn, kvnrm, topi, attn);

  // ---- sliding window attention (adds into attn) ----
  conv(w_k, wk, kHID * kD);
  conv(w_v, wv, kHID * kD);
  gemm(hbf, wk, swk0, kT, kD, kHID, kHID, kD, kD);
  gemm(hbf, wv, swv,  kT, kD, kHID, kHID, kD, kD);
  k_rms_rope<<<dim3(kT), dim3(128), 0, stream>>>(swk0, k_norm_w, kbf);
  conv(swv, vbf, kT * kD);
  sw_attn<<<dim3(kT / 16, kH), dim3(32), 0, stream>>>(qbf, kbf, vbf, sink_lg, attn);

  // ---- output projection ----
  conv(attn, attnbf, kT * kH * kD);
  conv(group_w, gw, kG * 512 * kDG);
  for (int g = 0; g < kG; ++g) {
    gemm(attnbf + g * 512, gw + (size_t)g * 512 * kDG, og + g * kDG,
         kT, kDG, 512, kH * kD, kDG, kG * kDG);
  }
  conv(og, ogbf, kT * kG * kDG);
  conv(final_w, fw, kG * kDG * kHID);
  gemm(ogbf, fw, (float*)d_out, kT, kHID, kG * kDG, kG * kDG, kHID, kHID);
}
